// Policy_55594056679770
// MI455X (gfx1250) — 8-bit matrix operand order repaired and checked by reading, compiled, not run on hardware
//
#include <hip/hip_runtime.h>
#include <hip/hip_bf16.h>

#define M_MACH   50
#define N_JOBS   100
#define MT_BITS  4
#define JS_BITS  2
#define NF       5400
#define NA       32
#define NC       1000
#define NC_PAD   1024
#define ROW_B    (2*NF)
#define ROW_CH   (ROW_B/16)
#define NCLAUSES (NA*NC)

typedef unsigned int u32x4 __attribute__((ext_vector_type(4)));
typedef int          v8i   __attribute__((ext_vector_type(8)));

#define WS_ADJ   0
#define WS_SIGN  16384
#define WS_OUT   (16384 + NA*NC_PAD)

__global__ void encode_kernel(const float* __restrict__ state,
                              const signed char* __restrict__ sign,
                              unsigned char* __restrict__ adj,
                              signed char* __restrict__ sign_pad,
                              signed char* __restrict__ out_pad) {
    __shared__ float smax;
    if (threadIdx.x == 0) {
        float m = 0.0f;
        for (int i = 0; i < M_MACH; ++i) m = fmaxf(m, state[i]);
        smax = m;
    }
    __syncthreads();
    const float mx = smax;

    for (int f = threadIdx.x; f < NF; f += blockDim.x) {
        int feat;
        if (f < M_MACH * MT_BITS) {
            int i = f >> 2, j = f & 3;
            int norm = (mx > 0.0f) ? (int)(state[i] / mx * 15.0f) : 0;
            feat = (norm >> j) & 1;
        } else if (f < M_MACH * MT_BITS + N_JOBS * JS_BITS) {
            int g = f - M_MACH * MT_BITS;
            int i = g >> 1, j = g & 1;
            int s = (int)(state[M_MACH + i] * 3.0f);
            if (s > 3) s = 3;
            feat = (s >> j) & 1;
        } else {
            feat = (int)state[M_MACH + N_JOBS + (f - (M_MACH*MT_BITS + N_JOBS*JS_BITS))];
        }
        adj[f]      = (feat == 0) ? 111 : 0;
        adj[NF + f] = (feat == 1) ? 111 : 0;
    }

    for (int idx = threadIdx.x; idx < NA * NC_PAD; idx += blockDim.x) {
        int a = idx >> 10, c = idx & (NC_PAD - 1);
        sign_pad[idx] = (c < NC) ? sign[a * NC + c] : (signed char)0;
        out_pad[idx]  = 0;
    }
}

__global__ void clause_kernel(const signed char* __restrict__ ta,
                              const unsigned char* __restrict__ adj,
                              signed char* __restrict__ out_pad) {
    __shared__ u32x4 sadj[ROW_CH];
    for (int i = threadIdx.x; i < ROW_CH; i += blockDim.x)
        sadj[i] = ((const u32x4*)adj)[i];
    __syncthreads();

    const int lane = threadIdx.x & 31;
    const int wave = blockIdx.x * (blockDim.x >> 5) + (threadIdx.x >> 5);
    if (wave >= NCLAUSES) return;

    const u32x4* row = (const u32x4*)(ta + (size_t)wave * ROW_B);

    unsigned int acc = 0u;
    for (int k = lane; k < ROW_CH; k += 32) {
        u32x4 t = __builtin_nontemporal_load(&row[k]);
        u32x4 a = sadj[k];
        acc |= (t.x + a.x) | (t.y + a.y) | (t.z + a.z) | (t.w + a.w);
    }
    for (int off = 16; off > 0; off >>= 1)
        acc |= (unsigned int)__shfl_xor((int)acc, off, 32);

    if (lane == 0) {
        int a = wave / NC, c = wave - a * NC;
        out_pad[a * NC_PAD + c] = (acc & 0x80808080u) ? 0 : 1;
    }
}

__global__ void qsum_wmma_kernel(const signed char* __restrict__ out_pad,
                                 const signed char* __restrict__ sign_pad,
                                 float* __restrict__ q) {
    const int lane = threadIdx.x & 31;
    const int g    = threadIdx.x >> 5;
    const int mn   = lane & 15;
    const int half = (lane >> 4) & 1;

    const int* __restrict__ ow = (const int*)out_pad;
    const int* __restrict__ sw = (const int*)sign_pad;

    v8i acc = {0, 0, 0, 0, 0, 0, 0, 0};

    for (int kb = 0; kb < NC_PAD / 64; ++kb) {
        const int base = (((g * 16 + mn) * NC_PAD) + kb * 64) >> 2;
        const int o = half ? 2 : 0;
        v8i A;
        A[0] = ow[base + o + 0];  A[1] = ow[base + o + 1];
        A[2] = ow[base + o + 4];  A[3] = ow[base + o + 5];
        A[4] = ow[base + o + 8];  A[5] = ow[base + o + 9];
        A[6] = ow[base + o + 12]; A[7] = ow[base + o + 13];

        const int bb = base + half * 2;
        v8i B;
        B[0] = sw[bb + 0]; B[1] = sw[bb + 1]; B[2] = sw[bb + 4]; B[3] = sw[bb + 5];
        B[4] = sw[bb + 8]; B[5] = sw[bb + 9]; B[6] = sw[bb + 12]; B[7] = sw[bb + 13];

        acc = __builtin_amdgcn_wmma_i32_16x16x64_iu8(false, A, true, B, acc,
                                                     false, false);
    }

    int diag = 0, agent = -1;
    if (lane < 8)        { diag = acc[lane];      agent = g * 16 + lane; }
    else if (lane >= 24) { diag = acc[lane - 24]; agent = g * 16 + lane - 16; }
    if (agent >= 0) q[agent] = (float)diag;
}

extern "C" void kernel_launch(void* const* d_in, const int* in_sizes, int n_in,
                              void* d_out, int out_size, void* d_ws, size_t ws_size,
                              hipStream_t stream) {
    (void)in_sizes; (void)n_in; (void)out_size; (void)ws_size;

    const float*       state = (const float*)d_in[0];
    const signed char* ta    = (const signed char*)d_in[1];
    const signed char* sign  = (const signed char*)d_in[2];
    float*             q     = (float*)d_out;

    unsigned char* ws       = (unsigned char*)d_ws;
    unsigned char* adj      = ws + WS_ADJ;
    signed char*   sign_pad = (signed char*)(ws + WS_SIGN);
    signed char*   out_pad  = (signed char*)(ws + WS_OUT);

    encode_kernel<<<1, 256, 0, stream>>>(state, sign, adj, sign_pad, out_pad);

    const int blocks = NCLAUSES / 8;
    clause_kernel<<<blocks, 256, 0, stream>>>(ta, adj, out_pad);

    qsum_wmma_kernel<<<1, 64, 0, stream>>>(out_pad, sign_pad, q);
}
